// Decoder_35802847379605
// MI455X (gfx1250) — matrix operand order repaired and checked by reading, not compiled, not run on hardware
//
#include <hip/hip_runtime.h>
#include <math.h>

// ---------------------------------------------------------------------------
// Tacotron2-style decoder on MI455X (gfx1250, wave32, WMMA).
//
// Heavy GEMMs (LSTM gates, processed_memory) run in bf16 via
// v_wmma_f32_16x16x32_bf16 with fp32 accumulate; recurrent state, biases,
// softmax and nonlinearities stay fp32. Weights are packed to bf16 once per
// call (~36MB) and stay L2-resident (192MB L2) across all 400 steps.
// GEMM tile order is mi-fast so the 4 waves sharing one weight (B) tile sit
// in the same workgroup and hit WGP$, cutting L2 weight traffic ~4x.
// Attention weights are staged to LDS via the gfx1250 async global->LDS path.
// ---------------------------------------------------------------------------

typedef __attribute__((ext_vector_type(8)))  float   v8f;
typedef __attribute__((ext_vector_type(8)))  __bf16  v8bf;
typedef __attribute__((ext_vector_type(16))) __bf16  v16bf;
typedef __attribute__((ext_vector_type(4)))  int     v4i;

#define BSZ   64
#define TIN   512
#define TOUTN 400
#define EDIM  512
#define ARNN  1024
#define DRNN  1024
#define PREN  256
#define ATTN  128
#define NFLT  32
#define KWIN  31
#define NMELS 80

#define KA (PREN + EDIM + ARNN)   /* 1792: [pre | ctx | ah]  */
#define KD (ARNN + EDIM + DRNN)   /* 2560: [ah | ctx | dh]   */
#define G4 (4 * ARNN)             /* 4096 gates              */

__device__ __forceinline__ unsigned short f2bf(float f) {
  unsigned u = __float_as_uint(f);
  unsigned r = ((u >> 16) & 1u) + 0x7FFFu;       // round to nearest even
  return (unsigned short)((u + r) >> 16);
}
__device__ __forceinline__ float sigm(float x) { return 1.0f / (1.0f + expf(-x)); }

// ---- gfx1250 async global->LDS staging (guarded; falls back to plain copy) -
typedef __attribute__((address_space(1))) v4i as1_v4i;   // global
typedef __attribute__((address_space(3))) v4i as3_v4i;   // LDS

#if __has_builtin(__builtin_amdgcn_global_load_async_to_lds_b128)
#define HAS_ASYNC_LDS 1
__device__ __forceinline__ void async_cp16(void* lds, const void* g) {
  __builtin_amdgcn_global_load_async_to_lds_b128(
      (as1_v4i*)g, (as3_v4i*)lds, 0, 0);
}
#else
#define HAS_ASYNC_LDS 0
#endif

__device__ __forceinline__ void wait_async0() {
#if __has_builtin(__builtin_amdgcn_s_wait_asynccnt)
  __builtin_amdgcn_s_wait_asynccnt(0);
#else
  asm volatile("s_wait_asynccnt 0" ::: "memory");
#endif
}

// ---------------- conversion / packing --------------------------------------

__global__ void cvt_f32_bf16(const float* __restrict__ src,
                             unsigned short* __restrict__ dst, int n) {
  int i = blockIdx.x * blockDim.x + threadIdx.x;
  if (i < n) dst[i] = f2bf(src[i]);
}

// pack [Wih | Whh] row-wise into one bf16 (G, Kx+Kh) matrix
__global__ void pack_w_bf16(const float* __restrict__ Wih,
                            const float* __restrict__ Whh,
                            unsigned short* __restrict__ out,
                            int Kx, int Kh, int G) {
  int K = Kx + Kh;
  long long i = (long long)blockIdx.x * blockDim.x + threadIdx.x;
  if (i >= (long long)G * K) return;
  int g = (int)(i / K), k = (int)(i % K);
  float v = (k < Kx) ? Wih[(size_t)g * Kx + k] : Whh[(size_t)g * Kh + (k - Kx)];
  out[i] = f2bf(v);
}

// pack three bf16 row blocks into one (BSZ, la+lb+lc) matrix (x_cat assembly)
__global__ void pack3_bf16(const unsigned short* __restrict__ a, int la,
                           const unsigned short* __restrict__ b, int lb,
                           const unsigned short* __restrict__ c, int lc,
                           unsigned short* __restrict__ out) {
  int K = la + lb + lc;
  int i = blockIdx.x * blockDim.x + threadIdx.x;
  if (i >= BSZ * K) return;
  int row = i / K, k = i % K;
  unsigned short v;
  if (k < la)            v = a[(size_t)row * la + k];
  else if (k < la + lb)  v = b[(size_t)row * lb + (k - la)];
  else                   v = c[(size_t)row * lc + (k - la - lb)];
  out[i] = v;
}

// ---------------- bf16 WMMA GEMM:  C[M,N] = A[M,K] * W[N,K]^T ---------------
// A row-major bf16, W row-major bf16 (weights as (N,K)), C row-major fp32.
// M%16==0, N%16==0, K%32==0.  One wave per 16x16 output tile, 8 waves/block.
// mi-fast tile order: waves that share a weight tile share a WGP (WGP$ reuse).
// Operand layouts follow CDNA5 ISA 7.12.2 (16-bit A 16x32, B 32x16, f32 C/D).

__global__ void __launch_bounds__(256)
gemm_bf16_wmma(const unsigned short* __restrict__ A,
               const unsigned short* __restrict__ W,
               float* __restrict__ C, int M, int N, int K) {
  int lane = threadIdx.x & 31;
  int wave = threadIdx.x >> 5;
  int mtiles = M >> 4, ntiles = N >> 4;
  int tile = blockIdx.x * 8 + wave;
  if (tile >= mtiles * ntiles) return;
  int mi = tile % mtiles;            // fast M: B tile shared within the block
  int ni = tile / mtiles;

  int hi = lane >> 4;        // half-wave select
  int l  = lane & 15;

  // A: lane holds row m = mi*16+l ; K chunks {base..base+7, base+16..base+23}
  const unsigned short* arow = A + (size_t)(mi * 16 + l) * K + hi * 8;
  // B: lane holds W row n = ni*16+l ; K chunks {base..base+7, base+16..base+23}
  const unsigned short* wrow = W + (size_t)(ni * 16 + l) * K + hi * 8;

  v8f acc = {};
  for (int k = 0; k < K; k += 32) {
    __builtin_prefetch(wrow + k + 1024, 0, 1);     // global_prefetch_b8, 2KB ahead
    v8bf a0 = *(const v8bf*)(arow + k);
    v8bf a1 = *(const v8bf*)(arow + k + 16);
    v16bf av = __builtin_shufflevector(a0, a1, 0, 1, 2, 3, 4, 5, 6, 7,
                                               8, 9, 10, 11, 12, 13, 14, 15);
    v8bf b0 = *(const v8bf*)(wrow + k);
    v8bf b1 = *(const v8bf*)(wrow + k + 16);
    v16bf bv = __builtin_shufflevector(b0, b1, 0, 1, 2, 3, 4, 5, 6, 7,
                                               8, 9, 10, 11, 12, 13, 14, 15);
    acc = __builtin_amdgcn_wmma_f32_16x16x32_bf16(
        false, av, false, bv, (short)0, acc, false, false);
  }
  // C/D layout: VGPR r -> row r (lanes 0-15) / row r+8 (lanes 16-31), col = l
  float* crow = C + (size_t)(mi * 16 + hi * 8) * N + ni * 16 + l;
#pragma unroll
  for (int r = 0; r < 8; ++r) crow[(size_t)r * N] = acc[r];
}

// ---------------- prenet (one-time precompute, 400x64x256) ------------------

__global__ void prenet1(const float* __restrict__ dec_in,   // (B,NMEL,TOUT)
                        const float* __restrict__ Wp1,      // (PRE,NMEL)
                        float* __restrict__ pre1) {         // (TOUT,B,PRE)
  long long idx = (long long)blockIdx.x * blockDim.x + threadIdx.x;
  if (idx >= (long long)TOUTN * BSZ * PREN) return;
  int s = (int)(idx / (BSZ * PREN));
  int r = (int)(idx % (BSZ * PREN));
  int b = r / PREN, j = r % PREN;
  float acc = 0.f;
  if (s > 0) {
    const float* x = dec_in + (size_t)b * NMELS * TOUTN + (s - 1);
    const float* w = Wp1 + (size_t)j * NMELS;
    for (int i = 0; i < NMELS; ++i) acc += w[i] * x[(size_t)i * TOUTN];
  }
  pre1[idx] = fmaxf(acc, 0.f);
}

__global__ void prenet2(const float* __restrict__ pre1,
                        const float* __restrict__ Wp2,      // (PRE,PRE)
                        unsigned short* __restrict__ pre_bf) {
  long long idx = (long long)blockIdx.x * blockDim.x + threadIdx.x;
  if (idx >= (long long)TOUTN * BSZ * PREN) return;
  int j = (int)(idx % PREN);
  const float* x = pre1 + (idx - j);
  const float* w = Wp2 + (size_t)j * PREN;
  float acc = 0.f;
  for (int i = 0; i < PREN; ++i) acc += w[i] * x[i];
  pre_bf[idx] = f2bf(fmaxf(acc, 0.f));
}

// ---------------- fused LSTM pointwise (i,f,g,o order) ----------------------

__global__ void lstm_pointwise(const float* __restrict__ z,     // (B,4H)
                               const float* __restrict__ bih,
                               const float* __restrict__ bhh,
                               float* __restrict__ h, float* __restrict__ c,
                               unsigned short* __restrict__ h_bf) {
  const int H = ARNN;
  int idx = blockIdx.x * blockDim.x + threadIdx.x;
  if (idx >= BSZ * H) return;
  int b = idx / H, j = idx % H;
  const float* zb = z + (size_t)b * 4 * H;
  float i_ = zb[j]         + bih[j]         + bhh[j];
  float f_ = zb[H + j]     + bih[H + j]     + bhh[H + j];
  float g_ = zb[2 * H + j] + bih[2 * H + j] + bhh[2 * H + j];
  float o_ = zb[3 * H + j] + bih[3 * H + j] + bhh[3 * H + j];
  float cn = sigm(f_) * c[idx] + sigm(i_) * tanhf(g_);
  float hn = sigm(o_) * tanhf(cn);
  c[idx] = cn;
  h[idx] = hn;
  h_bf[idx] = f2bf(hn);
}

// ---------------- location-sensitive attention (1 block / batch row) --------

__global__ void __launch_bounds__(256)
attention_step(const float* __restrict__ ah,        // (B,ARNN)
               const float* __restrict__ Wq,        // (ATT,ARNN)
               const float* __restrict__ Wconv,     // (NF,2,K)
               const float* __restrict__ Wloc,      // (ATT,NF)
               const float* __restrict__ Wv,        // (ATT)
               const float* __restrict__ pm,        // (B,TIN,ATT)
               const float* __restrict__ memory,    // (B,TIN,E)
               const unsigned char* __restrict__ mask,
               float* __restrict__ aw_g, float* __restrict__ awc_g,
               float* __restrict__ ctx, unsigned short* __restrict__ ctx_bf,
               float* __restrict__ out_align, int t_step) {
  __shared__ float s_aw[TIN], s_awc[TIN], s_pq[ATTN], s_en[TIN], s_red[256];
  int b = blockIdx.x, tid = threadIdx.x;

#if HAS_ASYNC_LDS
  // stage aw/awc through the gfx1250 async global->LDS path (ASYNCcnt)
  if (tid < 128)
    async_cp16(&s_aw[tid * 4], aw_g + (size_t)b * TIN + tid * 4);
  else
    async_cp16(&s_awc[(tid - 128) * 4], awc_g + (size_t)b * TIN + (tid - 128) * 4);
  wait_async0();
#else
  for (int i = tid; i < TIN; i += 256) {
    s_aw[i]  = aw_g[(size_t)b * TIN + i];
    s_awc[i] = awc_g[(size_t)b * TIN + i];
  }
#endif
  __syncthreads();

  if (tid < ATTN) {                                 // pq = ah @ Wq^T
    const float* hrow = ah + (size_t)b * ARNN;
    const float* wrow = Wq + (size_t)tid * ARNN;
    float s = 0.f;
    for (int h = 0; h < ARNN; ++h) s += hrow[h] * wrow[h];
    s_pq[tid] = s;
  }
  __syncthreads();

  const int pad = KWIN / 2;
  for (int p = 0; p < 2; ++p) {                     // energies, 2 pos / thread
    int t = tid + p * 256;
    float locf[NFLT];
#pragma unroll
    for (int f = 0; f < NFLT; ++f) locf[f] = 0.f;
    for (int k = 0; k < KWIN; ++k) {
      int tt = t + k - pad;
      if (tt < 0 || tt >= TIN) continue;
      float a0 = s_aw[tt], a1 = s_awc[tt];
      const float* w0 = Wconv + (size_t)k;          // [f][0][k], [f][1][k]
#pragma unroll
      for (int f = 0; f < NFLT; ++f)
        locf[f] += a0 * w0[(size_t)(f * 2) * KWIN] + a1 * w0[(size_t)(f * 2 + 1) * KWIN];
    }
    const float* pmrow = pm + ((size_t)b * TIN + t) * ATTN;
    float e = 0.f;
    for (int a = 0; a < ATTN; ++a) {
      float s = s_pq[a] + pmrow[a];
      const float* wl = Wloc + (size_t)a * NFLT;
#pragma unroll
      for (int f = 0; f < NFLT; ++f) s += locf[f] * wl[f];
      e += tanhf(s) * Wv[a];
    }
    if (mask[(size_t)b * TIN + t]) e = -INFINITY;
    s_en[t] = e;
  }
  __syncthreads();

  // softmax over TIN
  s_red[tid] = fmaxf(s_en[tid], s_en[tid + 256]);
  __syncthreads();
  for (int s = 128; s > 0; s >>= 1) {
    if (tid < s) s_red[tid] = fmaxf(s_red[tid], s_red[tid + s]);
    __syncthreads();
  }
  float mx = s_red[0];
  __syncthreads();
  float e0 = expf(s_en[tid] - mx), e1 = expf(s_en[tid + 256] - mx);
  s_red[tid] = e0 + e1;
  __syncthreads();
  for (int s = 128; s > 0; s >>= 1) {
    if (tid < s) s_red[tid] += s_red[tid + s];
    __syncthreads();
  }
  float inv = 1.0f / s_red[0];
  s_en[tid] = e0 * inv;
  s_en[tid + 256] = e1 * inv;
  __syncthreads();

  for (int p = 0; p < 2; ++p) {
    int t = tid + p * 256;
    float a_ = s_en[t];
    aw_g[(size_t)b * TIN + t] = a_;
    awc_g[(size_t)b * TIN + t] = s_awc[t] + a_;
    out_align[((size_t)b * TOUTN + t_step) * TIN + t] = a_;
  }

  // ctx[e] = sum_t aw[t] * memory[b,t,e]  (coalesced over e)
  for (int p = 0; p < 2; ++p) {
    int e = tid + p * 256;
    float acc = 0.f;
    const float* mrow = memory + (size_t)b * TIN * EDIM + e;
    for (int tt = 0; tt < TIN; ++tt)
      acc = fmaf(s_en[tt], mrow[(size_t)tt * EDIM], acc);
    ctx[(size_t)b * EDIM + e] = acc;
    ctx_bf[(size_t)b * EDIM + e] = f2bf(acc);
  }
}

// ---------------- projection + gate -----------------------------------------

__global__ void proj_step(const float* __restrict__ dh,    // (B,DRNN)
                          const float* __restrict__ ctx,   // (B,E)
                          const float* __restrict__ Wproj, // (NMEL, DRNN+E)
                          const float* __restrict__ bproj,
                          const float* __restrict__ Wgate, // (1, DRNN+E)
                          const float* __restrict__ bgate,
                          float* __restrict__ out, int t_step) {
  const int HC = DRNN + EDIM;
  int b = blockIdx.x, tid = threadIdx.x;
  if (tid > NMELS) return;
  const float* w = (tid < NMELS) ? (Wproj + (size_t)tid * HC) : Wgate;
  const float* hrow = dh + (size_t)b * DRNN;
  const float* crow = ctx + (size_t)b * EDIM;
  float acc = 0.f;
  for (int j = 0; j < DRNN; ++j) acc += w[j] * hrow[j];
  for (int j = 0; j < EDIM; ++j) acc += w[DRNN + j] * crow[j];
  const size_t melsz = (size_t)BSZ * NMELS * TOUTN;
  if (tid < NMELS)
    out[(size_t)b * NMELS * TOUTN + (size_t)tid * TOUTN + t_step] = acc + bproj[tid];
  else
    out[melsz + (size_t)b * TOUTN + t_step] = acc + bgate[0];
}

// ---------------------------------------------------------------------------

extern "C" void kernel_launch(void* const* d_in, const int* in_sizes, int n_in,
                              void* d_out, int out_size, void* d_ws, size_t ws_size,
                              hipStream_t stream) {
  const float* memory  = (const float*)d_in[0];
  const float* dec_in  = (const float*)d_in[1];
  const float* Wp1     = (const float*)d_in[2];
  const float* Wp2     = (const float*)d_in[3];
  const float* Wq      = (const float*)d_in[4];
  const float* Wm      = (const float*)d_in[5];
  const float* Wv      = (const float*)d_in[6];
  const float* Wconv   = (const float*)d_in[7];
  const float* Wloc    = (const float*)d_in[8];
  const float* Wih_a   = (const float*)d_in[9];
  const float* Whh_a   = (const float*)d_in[10];
  const float* bih_a   = (const float*)d_in[11];
  const float* bhh_a   = (const float*)d_in[12];
  const float* Wih_d   = (const float*)d_in[13];
  const float* Whh_d   = (const float*)d_in[14];
  const float* bih_d   = (const float*)d_in[15];
  const float* bhh_d   = (const float*)d_in[16];
  const float* Wproj   = (const float*)d_in[17];
  const float* bproj   = (const float*)d_in[18];
  const float* Wgate   = (const float*)d_in[19];
  const float* bgate   = (const float*)d_in[20];
  const unsigned char* mask = (const unsigned char*)d_in[21];
  float* out = (float*)d_out;

  // ---- workspace layout (256B aligned) ----
  size_t cur = 0;
  auto alloc = [&](size_t bytes) {
    size_t o = cur; cur += (bytes + 255) & ~(size_t)255; return o;
  };
  char* ws = (char*)d_ws;
  unsigned short* mem_bf  = (unsigned short*)(ws + alloc((size_t)BSZ * TIN * EDIM * 2));
  unsigned short* Wm_bf   = (unsigned short*)(ws + alloc((size_t)ATTN * EDIM * 2));
  float*          pmem    = (float*)(ws + alloc((size_t)BSZ * TIN * ATTN * 4));
  float*          pre1    = (float*)(ws + alloc((size_t)TOUTN * BSZ * PREN * 4));
  unsigned short* pre_bf  = (unsigned short*)(ws + alloc((size_t)TOUTN * BSZ * PREN * 2));
  unsigned short* Wa_bf   = (unsigned short*)(ws + alloc((size_t)G4 * KA * 2));
  unsigned short* Wd_bf   = (unsigned short*)(ws + alloc((size_t)G4 * KD * 2));
  unsigned short* xa      = (unsigned short*)(ws + alloc((size_t)BSZ * KA * 2));
  unsigned short* xd      = (unsigned short*)(ws + alloc((size_t)BSZ * KD * 2));
  float*          za      = (float*)(ws + alloc((size_t)BSZ * G4 * 4));
  float*          zd      = (float*)(ws + alloc((size_t)BSZ * G4 * 4));
  float*          ah      = (float*)(ws + alloc((size_t)BSZ * ARNN * 4));
  float*          ac      = (float*)(ws + alloc((size_t)BSZ * ARNN * 4));
  float*          dh      = (float*)(ws + alloc((size_t)BSZ * DRNN * 4));
  float*          dc      = (float*)(ws + alloc((size_t)BSZ * DRNN * 4));
  unsigned short* ah_bf   = (unsigned short*)(ws + alloc((size_t)BSZ * ARNN * 2));
  unsigned short* dh_bf   = (unsigned short*)(ws + alloc((size_t)BSZ * DRNN * 2));
  float*          ctx     = (float*)(ws + alloc((size_t)BSZ * EDIM * 4));
  unsigned short* ctx_bf  = (unsigned short*)(ws + alloc((size_t)BSZ * EDIM * 2));
  float*          aw      = (float*)(ws + alloc((size_t)BSZ * TIN * 4));
  float*          awc     = (float*)(ws + alloc((size_t)BSZ * TIN * 4));
  (void)ws_size; (void)n_in; (void)in_sizes; (void)out_size;

  const size_t melsz = (size_t)BSZ * NMELS * TOUTN;
  const size_t gatesz = (size_t)BSZ * TOUTN;
  float* out_align = out + melsz + gatesz;

  // ---- zero-init recurrent state ----
  (void)hipMemsetAsync(ah, 0, (size_t)BSZ * ARNN * 4, stream);
  (void)hipMemsetAsync(ac, 0, (size_t)BSZ * ARNN * 4, stream);
  (void)hipMemsetAsync(dh, 0, (size_t)BSZ * DRNN * 4, stream);
  (void)hipMemsetAsync(dc, 0, (size_t)BSZ * DRNN * 4, stream);
  (void)hipMemsetAsync(ah_bf, 0, (size_t)BSZ * ARNN * 2, stream);
  (void)hipMemsetAsync(dh_bf, 0, (size_t)BSZ * DRNN * 2, stream);
  (void)hipMemsetAsync(ctx, 0, (size_t)BSZ * EDIM * 4, stream);
  (void)hipMemsetAsync(ctx_bf, 0, (size_t)BSZ * EDIM * 2, stream);
  (void)hipMemsetAsync(aw, 0, (size_t)BSZ * TIN * 4, stream);
  (void)hipMemsetAsync(awc, 0, (size_t)BSZ * TIN * 4, stream);

  // ---- one-time precompute ----
  {
    int n = BSZ * TIN * EDIM;
    cvt_f32_bf16<<<(n + 255) / 256, 256, 0, stream>>>(memory, mem_bf, n);
    n = ATTN * EDIM;
    cvt_f32_bf16<<<(n + 255) / 256, 256, 0, stream>>>(Wm, Wm_bf, n);
    long long na = (long long)G4 * KA;
    pack_w_bf16<<<(unsigned)((na + 255) / 256), 256, 0, stream>>>(
        Wih_a, Whh_a, Wa_bf, PREN + EDIM, ARNN, G4);
    long long nd = (long long)G4 * KD;
    pack_w_bf16<<<(unsigned)((nd + 255) / 256), 256, 0, stream>>>(
        Wih_d, Whh_d, Wd_bf, ARNN + EDIM, DRNN, G4);
    // processed_memory: (B*TIN, ATT) = mem_bf (32768,512) @ Wm_bf(128,512)^T
    int tiles = (BSZ * TIN / 16) * (ATTN / 16);
    gemm_bf16_wmma<<<(tiles + 7) / 8, 256, 0, stream>>>(
        mem_bf, Wm_bf, pmem, BSZ * TIN, ATTN, EDIM);
    long long np = (long long)TOUTN * BSZ * PREN;
    prenet1<<<(unsigned)((np + 255) / 256), 256, 0, stream>>>(dec_in, Wp1, pre1);
    prenet2<<<(unsigned)((np + 255) / 256), 256, 0, stream>>>(pre1, Wp2, pre_bf);
  }

  // ---- 400 sequential decoder steps ----
  const int tiles_a = (BSZ / 16) * (G4 / 16);   // 1024 tiles
  for (int t = 0; t < TOUTN; ++t) {
    // x_cat for attention LSTM: [pre_t | ctx | ah]
    pack3_bf16<<<(BSZ * KA + 255) / 256, 256, 0, stream>>>(
        pre_bf + (size_t)t * BSZ * PREN, PREN, ctx_bf, EDIM, ah_bf, ARNN, xa);
    gemm_bf16_wmma<<<(tiles_a + 7) / 8, 256, 0, stream>>>(xa, Wa_bf, za, BSZ, G4, KA);
    lstm_pointwise<<<(BSZ * ARNN + 255) / 256, 256, 0, stream>>>(
        za, bih_a, bhh_a, ah, ac, ah_bf);

    attention_step<<<BSZ, 256, 0, stream>>>(
        ah, Wq, Wconv, Wloc, Wv, pmem, memory, mask,
        aw, awc, ctx, ctx_bf, out_align, t);

    // x_cat for decoder LSTM: [ah | ctx | dh]
    pack3_bf16<<<(BSZ * KD + 255) / 256, 256, 0, stream>>>(
        ah_bf, ARNN, ctx_bf, EDIM, dh_bf, DRNN, xd);
    gemm_bf16_wmma<<<(tiles_a + 7) / 8, 256, 0, stream>>>(xd, Wd_bf, zd, BSZ, G4, KD);
    lstm_pointwise<<<(BSZ * DRNN + 255) / 256, 256, 0, stream>>>(
        zd, bih_d, bhh_d, dh, dc, dh_bf);

    proj_step<<<BSZ, 96, 0, stream>>>(dh, ctx, Wproj, bproj, Wgate, bgate, out, t);
  }
}
